// SelfAttention_87522843558152
// MI455X (gfx1250) — matrix operand order repaired and checked by reading, not compiled, not run on hardware
//
#include <hip/hip_runtime.h>
#include <hip/hip_bf16.h>

// ---------------- types ----------------
typedef _Float16 v16h __attribute__((ext_vector_type(16)));
typedef _Float16 v8h  __attribute__((ext_vector_type(8)));
typedef float    v8f  __attribute__((ext_vector_type(8)));

static __device__ __forceinline__ v8h ld8(const _Float16* p) {
    return *(const v8h*)p;
}
static __device__ __forceinline__ v16h cat8(v8h lo, v8h hi) {
    v16h r;
#pragma unroll
    for (int i = 0; i < 8; ++i) { r[i] = lo[i]; r[8 + i] = hi[i]; }
    return r;
}
static __device__ __forceinline__ v8f wmma_f16(v16h a, v16h b, v8f c) {
    // D = A(16x32 f16) * B(32x16 f16) + C(16x16 f32)
    return __builtin_amdgcn_wmma_f32_16x16x32_f16(
        false, a, false, b, (short)0, c, false, false);
}

// CDNA5 async global->LDS copy (16B per lane), tracked by ASYNCcnt.
static __device__ __forceinline__ void async_ld_b128(unsigned lds_off,
                                                     const _Float16* gptr) {
    asm volatile("global_load_async_to_lds_b128 %0, %1, off"
                 :: "v"(lds_off), "v"((unsigned long long)(uintptr_t)gptr)
                 : "memory");
}
static __device__ __forceinline__ void wait_async0() {
    asm volatile("s_wait_asynccnt 0" ::: "memory");
}

// Problem constants
#define S_LEN 4096
#define D_DIM 768
#define H_NUM 12
#define DK_DIM 64
#define LDSTR 72   // LDS row stride in halves: 144B (16B aligned), 36-dword bank shift

// ---------------- kernel 1: fp32 -> f16 convert ----------------
__global__ void cvt_f32_to_f16(const float* __restrict__ src,
                               _Float16* __restrict__ dst, int n) {
    int i = blockIdx.x * blockDim.x + threadIdx.x;
    if (i < n) dst[i] = (_Float16)src[i];
}

// ---------------- kernel 2: QKV projection (y = x W^T + b) ----------------
// 64x64 block tile, 4 waves; W panel async-staged into double-buffered LDS,
// one barrier per k-step, compute on buf[p] overlaps staging of buf[1-p].
__global__ __launch_bounds__(128) void qkv_proj(
    const _Float16* __restrict__ xh,
    const _Float16* __restrict__ Wqh, const _Float16* __restrict__ Wkh,
    const _Float16* __restrict__ Wvh,
    const float* __restrict__ bq, const float* __restrict__ bk,
    const float* __restrict__ bv,
    _Float16* __restrict__ Qh, _Float16* __restrict__ Kh,
    _Float16* __restrict__ Vh) {
    __shared__ _Float16 Wsm[2][64 * 40];   // [buf][o_local*40 + k32]

    const int mat  = blockIdx.x / (64 * 12);          // 0=Q 1=K 2=V
    const int rem  = blockIdx.x % (64 * 12);
    const int qt64 = rem / 12;
    const int ot64 = rem % 12;

    const _Float16* W    = (mat == 0) ? Wqh : (mat == 1) ? Wkh : Wvh;
    const float*    bias = (mat == 0) ? bq  : (mat == 1) ? bk  : bv;
    _Float16*       dst  = (mat == 0) ? Qh  : (mat == 1) ? Kh  : Vh;

    const int lane = threadIdx.x & 31;
    const int wave = threadIdx.x >> 5;
    const int n    = lane & 15;
    const int hi   = lane >> 4;

    const int row0 = qt64 * 64 + wave * 16;
    const _Float16* arow = xh + (row0 + n) * D_DIM;

    // staging map: thread t -> o-row = t/2 (0..63), 32-half half = t&1
    const int orow = threadIdx.x >> 1;
    const int hh   = threadIdx.x & 1;
    const _Float16* wsrc = W + (ot64 * 64 + orow) * D_DIM + hh * 16;
    unsigned wlds[2];
    wlds[0] = (unsigned)(uintptr_t)&Wsm[0][orow * 40 + hh * 16];
    wlds[1] = (unsigned)(uintptr_t)&Wsm[1][orow * 40 + hh * 16];

    // prologue: stage k-step 0 into buf 0
    async_ld_b128(wlds[0],      wsrc);
    async_ld_b128(wlds[0] + 16, wsrc + 8);
    wait_async0();

    v8f c0 = {}, c1 = {}, c2 = {}, c3 = {};
    int p = 0;
    for (int ki = 0; ki < 24; ++ki) {
        __syncthreads();                          // buf[p] visible to all
        // stage next k-step into buf[1-p] (wraps; redundant last iter)
        const int kn = ((ki + 1) % 24) * 32;
        async_ld_b128(wlds[1 - p],      wsrc + kn);
        async_ld_b128(wlds[1 - p] + 16, wsrc + kn + 8);

        const int k0 = ki * 32;
        v16h a = cat8(ld8(arow + k0 + hi * 16),
                      ld8(arow + k0 + hi * 16 + 8));
        const _Float16* wb = Wsm[p];
        // issue all B-fragment loads first, then the WMMA group
        v16h b0 = cat8(ld8(&wb[(0 * 16 + n) * 40 + hi * 16]),
                       ld8(&wb[(0 * 16 + n) * 40 + hi * 16 + 8]));
        v16h b1 = cat8(ld8(&wb[(1 * 16 + n) * 40 + hi * 16]),
                       ld8(&wb[(1 * 16 + n) * 40 + hi * 16 + 8]));
        v16h b2 = cat8(ld8(&wb[(2 * 16 + n) * 40 + hi * 16]),
                       ld8(&wb[(2 * 16 + n) * 40 + hi * 16 + 8]));
        v16h b3 = cat8(ld8(&wb[(3 * 16 + n) * 40 + hi * 16]),
                       ld8(&wb[(3 * 16 + n) * 40 + hi * 16 + 8]));
        c0 = wmma_f16(a, b0, c0);
        c1 = wmma_f16(a, b1, c1);
        c2 = wmma_f16(a, b2, c2);
        c3 = wmma_f16(a, b3, c3);

        wait_async0();                            // buf[1-p] staged
        p ^= 1;
    }

#pragma unroll
    for (int ot = 0; ot < 4; ++ot) {
        const v8f& c = (ot == 0) ? c0 : (ot == 1) ? c1 : (ot == 2) ? c2 : c3;
        const int o  = ot64 * 64 + ot * 16 + n;
        const float bb = bias[o];
        _Float16* outp = dst + (o >> 6) * (S_LEN * DK_DIM) + (o & 63);
#pragma unroll
        for (int r = 0; r < 8; ++r) {
            const int s = row0 + r + hi * 8;      // C layout: M = r + hi*8
            outp[s * DK_DIM] = (_Float16)(c[r] + bb);
        }
    }
}

// ---------------- kernel 3: flash attention, 64-wide KV, pipelined ----------------
// grid = H * (S/64), 128 threads (4 waves). Wave w: 16 query rows, DK=64 out.
// K async-staged, V loaded early / scattered after compute; double-buffered LDS,
// one barrier per KV block.
__global__ __launch_bounds__(128) void attn(
    const _Float16* __restrict__ Qh, const _Float16* __restrict__ Kh,
    const _Float16* __restrict__ Vh, float* __restrict__ out) {
    __shared__ _Float16 Ksm[2][64 * LDSTR];      // [buf][kv_local][d]
    __shared__ _Float16 Vsm[2][64 * LDSTR];      // [buf] transposed [d][kv_local]
    __shared__ _Float16 Psm[4][16 * LDSTR];      // per-wave P tile [M][kv_local]

    const int h    = blockIdx.x / (S_LEN / 64);
    const int qb   = blockIdx.x % (S_LEN / 64);
    const int lane = threadIdx.x & 31;
    const int wave = threadIdx.x >> 5;
    const int n    = lane & 15;
    const int hi   = lane >> 4;
    const int q0   = qb * 64 + wave * 16;

    const _Float16* Qhead = Qh + h * (S_LEN * DK_DIM);
    const _Float16* Khead = Kh + h * (S_LEN * DK_DIM);
    const _Float16* Vhead = Vh + h * (S_LEN * DK_DIM);

    // Q A-fragments (two 32-wide K chunks), loaded once.
    const _Float16* qrow = Qhead + (q0 + n) * DK_DIM;
    const v16h qa0 = cat8(ld8(qrow +      hi * 16), ld8(qrow +      hi * 16 + 8));
    const v16h qa1 = cat8(ld8(qrow + 32 + hi * 16), ld8(qrow + 32 + hi * 16 + 8));

    v8f z0 = {}, z1 = {}, z2 = {}, z3 = {};
    float m[8], l[8];
#pragma unroll
    for (int r = 0; r < 8; ++r) { m[r] = -1e30f; l[r] = 0.0f; }
    const float scale = 0.125f;   // 1/sqrt(64)

    // staging map: thread t -> kv row = t/2 (0..63), 32-half half = t&1
    const int rrow = threadIdx.x >> 1;
    const int hh   = threadIdx.x & 1;
    const _Float16* kstg = Khead + rrow * DK_DIM + hh * 32;
    const _Float16* vstg = Vhead + rrow * DK_DIM + hh * 32;
    unsigned klds[2];
    klds[0] = (unsigned)(uintptr_t)&Ksm[0][rrow * LDSTR + hh * 32];
    klds[1] = (unsigned)(uintptr_t)&Ksm[1][rrow * LDSTR + hh * 32];

    // prologue: stage KV block 0 into buf 0
    {
        async_ld_b128(klds[0],      kstg);
        async_ld_b128(klds[0] + 16, kstg + 8);
        async_ld_b128(klds[0] + 32, kstg + 16);
        async_ld_b128(klds[0] + 48, kstg + 24);
        v8h va = ld8(vstg), vb_ = ld8(vstg + 8), vc = ld8(vstg + 16), vd = ld8(vstg + 24);
        _Float16* vt = &Vsm[0][rrow];
#pragma unroll
        for (int i = 0; i < 8; ++i) {
            vt[(hh * 32 + i)      * LDSTR] = va[i];
            vt[(hh * 32 + 8 + i)  * LDSTR] = vb_[i];
            vt[(hh * 32 + 16 + i) * LDSTR] = vc[i];
            vt[(hh * 32 + 24 + i) * LDSTR] = vd[i];
        }
        wait_async0();
    }

    int p = 0;
    for (int it = 0; it < S_LEN / 64; ++it) {
        __syncthreads();                          // buf[p] visible to all
        // ---- issue next block's loads (wraps; redundant last iter) ----
        const int kvn = ((it + 1) & (S_LEN / 64 - 1)) * 64;
        async_ld_b128(klds[1 - p],      kstg + kvn * DK_DIM);
        async_ld_b128(klds[1 - p] + 16, kstg + kvn * DK_DIM + 8);
        async_ld_b128(klds[1 - p] + 32, kstg + kvn * DK_DIM + 16);
        async_ld_b128(klds[1 - p] + 48, kstg + kvn * DK_DIM + 24);
        v8h va = ld8(vstg + kvn * DK_DIM);
        v8h vb_ = ld8(vstg + kvn * DK_DIM + 8);
        v8h vc = ld8(vstg + kvn * DK_DIM + 16);
        v8h vd = ld8(vstg + kvn * DK_DIM + 24);

        // ---- scores on buf[p]: four 16x16 tiles (loads grouped before WMMA)
        const _Float16* kbuf = Ksm[p];
        v8f s0 = {}, s1 = {}, s2 = {}, s3 = {};
        {
            const _Float16* kb0 = &kbuf[(0 * 16 + n) * LDSTR];
            const _Float16* kb1 = &kbuf[(1 * 16 + n) * LDSTR];
            v16h b0A = cat8(ld8(kb0 + hi * 16),      ld8(kb0 + hi * 16 + 8));
            v16h b0B = cat8(ld8(kb0 + 32 + hi * 16), ld8(kb0 + 32 + hi * 16 + 8));
            v16h b1A = cat8(ld8(kb1 + hi * 16),      ld8(kb1 + hi * 16 + 8));
            v16h b1B = cat8(ld8(kb1 + 32 + hi * 16), ld8(kb1 + 32 + hi * 16 + 8));
            s0 = wmma_f16(qa0, b0A, s0);
            s0 = wmma_f16(qa1, b0B, s0);
            s1 = wmma_f16(qa0, b1A, s1);
            s1 = wmma_f16(qa1, b1B, s1);
            const _Float16* kb2 = &kbuf[(2 * 16 + n) * LDSTR];
            const _Float16* kb3 = &kbuf[(3 * 16 + n) * LDSTR];
            v16h b2A = cat8(ld8(kb2 + hi * 16),      ld8(kb2 + hi * 16 + 8));
            v16h b2B = cat8(ld8(kb2 + 32 + hi * 16), ld8(kb2 + 32 + hi * 16 + 8));
            v16h b3A = cat8(ld8(kb3 + hi * 16),      ld8(kb3 + hi * 16 + 8));
            v16h b3B = cat8(ld8(kb3 + 32 + hi * 16), ld8(kb3 + 32 + hi * 16 + 8));
            s2 = wmma_f16(qa0, b2A, s2);
            s2 = wmma_f16(qa1, b2B, s2);
            s3 = wmma_f16(qa0, b3A, s3);
            s3 = wmma_f16(qa1, b3B, s3);
        }

        // ---- online softmax over 64 columns ----
        _Float16* pw = &Psm[wave][0];
#pragma unroll
        for (int r = 0; r < 8; ++r) {
            float bm = fmaxf(fmaxf(s0[r], s1[r]), fmaxf(s2[r], s3[r]));
#pragma unroll
            for (int off = 1; off < 16; off <<= 1)
                bm = fmaxf(bm, __shfl_xor(bm, off, 16));
            bm *= scale;
            const float mn    = fmaxf(m[r], bm);
            const float alpha = __expf(m[r] - mn);
            const float p0 = __expf(s0[r] * scale - mn);
            const float p1 = __expf(s1[r] * scale - mn);
            const float p2 = __expf(s2[r] * scale - mn);
            const float p3 = __expf(s3[r] * scale - mn);
            float rs = (p0 + p1) + (p2 + p3);
#pragma unroll
            for (int off = 1; off < 16; off <<= 1)
                rs += __shfl_xor(rs, off, 16);
            m[r] = mn;
            l[r] = l[r] * alpha + rs;
            z0[r] *= alpha; z1[r] *= alpha; z2[r] *= alpha; z3[r] *= alpha;
            const int M = r + hi * 8;                 // C-layout row
            pw[M * LDSTR +      n] = (_Float16)p0;
            pw[M * LDSTR + 16 + n] = (_Float16)p1;
            pw[M * LDSTR + 32 + n] = (_Float16)p2;
            pw[M * LDSTR + 48 + n] = (_Float16)p3;
        }

        // ---- P back in A-layout (per-wave LDS; same-wave DS ops in order)
        const _Float16* pr = &pw[n * LDSTR + hi * 16];
        v16h pa0 = cat8(ld8(pr),      ld8(pr + 8));
        v16h pa1 = cat8(ld8(pr + 32), ld8(pr + 40));

        // ---- z += P * V on buf[p] (loads grouped before WMMA) ----
        const _Float16* vbuf = Vsm[p];
        {
            const _Float16* vb0 = &vbuf[(0 * 16 + n) * LDSTR];
            const _Float16* vb1 = &vbuf[(1 * 16 + n) * LDSTR];
            v16h v0A = cat8(ld8(vb0 + hi * 16),      ld8(vb0 + hi * 16 + 8));
            v16h v0B = cat8(ld8(vb0 + 32 + hi * 16), ld8(vb0 + 32 + hi * 16 + 8));
            v16h v1A = cat8(ld8(vb1 + hi * 16),      ld8(vb1 + hi * 16 + 8));
            v16h v1B = cat8(ld8(vb1 + 32 + hi * 16), ld8(vb1 + 32 + hi * 16 + 8));
            z0 = wmma_f16(pa0, v0A, z0);
            z0 = wmma_f16(pa1, v0B, z0);
            z1 = wmma_f16(pa0, v1A, z1);
            z1 = wmma_f16(pa1, v1B, z1);
            const _Float16* vb2 = &vbuf[(2 * 16 + n) * LDSTR];
            const _Float16* vb3 = &vbuf[(3 * 16 + n) * LDSTR];
            v16h v2A = cat8(ld8(vb2 + hi * 16),      ld8(vb2 + hi * 16 + 8));
            v16h v2B = cat8(ld8(vb2 + 32 + hi * 16), ld8(vb2 + 32 + hi * 16 + 8));
            v16h v3A = cat8(ld8(vb3 + hi * 16),      ld8(vb3 + hi * 16 + 8));
            v16h v3B = cat8(ld8(vb3 + 32 + hi * 16), ld8(vb3 + 32 + hi * 16 + 8));
            z2 = wmma_f16(pa0, v2A, z2);
            z2 = wmma_f16(pa1, v2B, z2);
            z3 = wmma_f16(pa0, v3A, z3);
            z3 = wmma_f16(pa1, v3B, z3);
        }

        // ---- scatter next block's V (transposed) into buf[1-p] ----
        {
            _Float16* vt = &Vsm[1 - p][rrow];
#pragma unroll
            for (int i = 0; i < 8; ++i) {
                vt[(hh * 32 + i)      * LDSTR] = va[i];
                vt[(hh * 32 + 8 + i)  * LDSTR] = vb_[i];
                vt[(hh * 32 + 16 + i) * LDSTR] = vc[i];
                vt[(hh * 32 + 24 + i) * LDSTR] = vd[i];
            }
        }
        wait_async0();                            // next K block staged
        p ^= 1;
    }

    // ---- normalize and store fp32 into merged [S][H*DK] layout ----
#pragma unroll
    for (int r = 0; r < 8; ++r) {
        const float inv = 1.0f / l[r];
        const int   s   = q0 + r + hi * 8;
        float* op = out + s * D_DIM + h * DK_DIM + n;
        op[0]  = z0[r] * inv;
        op[16] = z1[r] * inv;
        op[32] = z2[r] * inv;
        op[48] = z3[r] * inv;
    }
}

// ---------------- host side ----------------
extern "C" void kernel_launch(void* const* d_in, const int* in_sizes, int n_in,
                              void* d_out, int out_size, void* d_ws, size_t ws_size,
                              hipStream_t stream) {
    const float* x  = (const float*)d_in[0];
    const float* Wq = (const float*)d_in[1];
    const float* bq = (const float*)d_in[2];
    const float* Wk = (const float*)d_in[3];
    const float* bk = (const float*)d_in[4];
    const float* Wv = (const float*)d_in[5];
    const float* bv = (const float*)d_in[6];
    float* out = (float*)d_out;

    _Float16* ws = (_Float16*)d_ws;
    const size_t N_X = (size_t)S_LEN * D_DIM;      // 3,145,728
    const size_t N_W = (size_t)D_DIM * D_DIM;      //   589,824
    _Float16* xh  = ws;
    _Float16* wqh = xh  + N_X;
    _Float16* wkh = wqh + N_W;
    _Float16* wvh = wkh + N_W;
    _Float16* Qh  = wvh + N_W;
    _Float16* Kh  = Qh + N_X;
    _Float16* Vh  = Kh + N_X;
    // total: 4*N_X + 3*N_W halves = ~28.7 MB of workspace

    cvt_f32_to_f16<<<(int)(N_X / 256), 256, 0, stream>>>(x,  xh,  (int)N_X);
    cvt_f32_to_f16<<<(int)(N_W / 256), 256, 0, stream>>>(Wq, wqh, (int)N_W);
    cvt_f32_to_f16<<<(int)(N_W / 256), 256, 0, stream>>>(Wk, wkh, (int)N_W);
    cvt_f32_to_f16<<<(int)(N_W / 256), 256, 0, stream>>>(Wv, wvh, (int)N_W);

    // 3 mats * 64 row-blocks * 12 col-blocks = 2304 WGs
    qkv_proj<<<2304, 128, 0, stream>>>(xh, wqh, wkh, wvh, bq, bk, bv,
                                       Qh, Kh, Vh);

    // H * (S/64) = 12 * 64 = 768 WGs
    attn<<<H_NUM * (S_LEN / 64), 128, 0, stream>>>(Qh, Kh, Vh, out);
}
